// MoEFFN_78675210928554
// MI455X (gfx1250) — matrix operand order repaired and checked by reading, not compiled, not run on hardware
//
#include <hip/hip_runtime.h>
#include <hip/hip_bf16.h>
#include <math.h>

#define D_MODEL 512
#define D_FF    2048
#define NE      16
#define BTOK    2048
#define CAP     320
#define SLOTS   (NE * CAP)   // 5120

#define BLK_M   64
#define BLK_N   128
#define PAD_ROW 40           // 32 k-values + 8 pad (80B row) -> conflict-free b128

typedef __bf16 v16bf __attribute__((ext_vector_type(16)));
typedef __bf16 v8bf  __attribute__((ext_vector_type(8)));
typedef float  v8f   __attribute__((ext_vector_type(8)));

// ---------------------------------------------------------------- routing ---

__global__ void moe_init_kernel(int* tok_of_slot, int* used, int* token_slots) {
    int i = blockIdx.x * blockDim.x + threadIdx.x;
    if (i < SLOTS)    tok_of_slot[i] = -1;
    if (i < BTOK)     used[i] = 0;
    if (i < 2 * BTOK) token_slots[i] = -1;
}

// One thread per expert; serial token scan reproduces the reference's
// exclusive-cumsum capacity gate exactly.
__global__ void moe_route_kernel(const int* __restrict__ routes,
                                 int* __restrict__ tok_of_slot,
                                 int* __restrict__ used,
                                 int* __restrict__ token_slots) {
    int e = threadIdx.x;
    if (e >= NE) return;
    int cum = 0;
    for (int t = 0; t < BTOK; ++t) {
        int e0 = (routes[2 * t]     * NE) / BTOK;
        int e1 = (routes[2 * t + 1] * NE) / BTOK;
        bool sel = (e0 == e) || (e1 == e);
        if (sel) {
            if (cum < CAP) {
                int slot = e * CAP + cum;
                tok_of_slot[slot] = t;
                int pos = atomicAdd(&used[t], 1);
                token_slots[2 * t + pos] = slot;
            }
            ++cum;
        }
    }
}

__global__ void moe_gather_kernel(const float* __restrict__ x,
                                  const int* __restrict__ tok_of_slot,
                                  float* __restrict__ xg) {
    int i = blockIdx.x * blockDim.x + threadIdx.x;
    int slot = i >> 9;          // /512
    int d    = i & 511;
    int tok  = tok_of_slot[slot];
    xg[i] = (tok >= 0) ? x[(size_t)tok * D_MODEL + d] : 0.0f;
}

// ------------------------------------------------------------ WMMA GEMM -----
// C[M,N] = act(A[M,K] @ B[K,N] + bias), bf16x3 split (ah*bh + al*bh + ah*bl,
// f32 accumulate). Block = 64x128 tile, 8 waves (4m x 2n), wave = 16x64 strip.
// A/B panels are split to bf16 hi/lo once per block in LDS, stored in WMMA
// fragment order so waves fetch operands with contiguous ds_load_b128.

__device__ __forceinline__ v16bf cat8(v8bf a, v8bf b) {
    return __builtin_shufflevector(a, b, 0, 1, 2, 3, 4, 5, 6, 7,
                                         8, 9, 10, 11, 12, 13, 14, 15);
}

__global__ __launch_bounds__(256)
void moe_wmma_gemm_kernel(const float* __restrict__ A, long long strideAe,
                          const float* __restrict__ B, long long strideBe,
                          const float* __restrict__ bias, long long strideBiasE,
                          float* __restrict__ C, long long strideCe,
                          int N, int K, int nBlks, int applyGelu) {
    __shared__ __attribute__((aligned(16))) __bf16 Ah[BLK_M * PAD_ROW];
    __shared__ __attribute__((aligned(16))) __bf16 Al[BLK_M * PAD_ROW];
    __shared__ __attribute__((aligned(16))) __bf16 Bh[BLK_N * PAD_ROW];
    __shared__ __attribute__((aligned(16))) __bf16 Bl[BLK_N * PAD_ROW];

    const int e = blockIdx.z;
    const float* Ae = A + (size_t)e * strideAe;
    const float* Be = B + (size_t)e * strideBe;
    const float* be = bias + (size_t)e * strideBiasE;
    float* Ce = C + (size_t)e * strideCe;

    const int tid  = threadIdx.x;
    const int lane = tid & 31;
    const int wave = tid >> 5;
    const int wm   = wave >> 1;            // 0..3 : 16-row strip in block
    const int wn   = wave & 1;             // 0..1 : 64-col strip in block
    const int mBlk = blockIdx.x / nBlks;
    const int nBlk = blockIdx.x % nBlks;
    const int m0   = mBlk * BLK_M;
    const int n0   = nBlk * BLK_N;

    // staging assignments (all 256 threads)
    const int aRow = tid >> 2;             // 0..63
    const int aK   = (tid & 3) * 8;        // 0,8,16,24
    const int bN   = tid & 127;            // 0..127
    const int bKh  = (tid >> 7) * 16;      // 0,16

    const int laneM = lane & 15;
    const int half  = lane >> 4;

    const v8f vzero = {0.f, 0.f, 0.f, 0.f, 0.f, 0.f, 0.f, 0.f};
    v8f acc[4] = {vzero, vzero, vzero, vzero};

    for (int k0 = 0; k0 < K; k0 += 32) {
        // ---- stage A panel: 64 rows x 32 k, hi/lo split, fragment layout ----
        {
            const float* src = Ae + (size_t)(m0 + aRow) * K + k0 + aK;
            const float4 q0 = *(const float4*)(src);
            const float4 q1 = *(const float4*)(src + 4);
            float v[8] = {q0.x, q0.y, q0.z, q0.w, q1.x, q1.y, q1.z, q1.w};
            v8bf hv, lv;
#pragma unroll
            for (int j = 0; j < 8; ++j) {
                __bf16 h = (__bf16)v[j];
                hv[j] = h;
                lv[j] = (__bf16)(v[j] - (float)h);
            }
            const int base = aRow * PAD_ROW + aK;
            *(v8bf*)(Ah + base) = hv;
            *(v8bf*)(Al + base) = lv;
        }
        // ---- stage B panel: 32 k x 128 n (coalesced rows), transposed into
        //      per-column fragment layout Bh[n][k] ----
        {
            const float* src = Be + (size_t)(k0 + bKh) * N + n0 + bN;
            float v[16];
#pragma unroll
            for (int j = 0; j < 16; ++j) v[j] = src[(size_t)j * N];
            v8bf hv0, lv0, hv1, lv1;
#pragma unroll
            for (int j = 0; j < 8; ++j) {
                __bf16 h0 = (__bf16)v[j];
                hv0[j] = h0;
                lv0[j] = (__bf16)(v[j] - (float)h0);
                __bf16 h1 = (__bf16)v[8 + j];
                hv1[j] = h1;
                lv1[j] = (__bf16)(v[8 + j] - (float)h1);
            }
            const int base = bN * PAD_ROW + bKh;
            *(v8bf*)(Bh + base)     = hv0;
            *(v8bf*)(Bh + base + 8) = hv1;
            *(v8bf*)(Bl + base)     = lv0;
            *(v8bf*)(Bl + base + 8) = lv1;
        }
        __syncthreads();

        // ---- fragments + WMMA ----
        {
            // A fragment (ISA 7.12.2): lane half h holds k = 8h+[0..7], 16+8h+[0..7]
            const int aoff = (wm * 16 + laneM) * PAD_ROW + half * 8;
            const v16bf ah = cat8(*(const v8bf*)(Ah + aoff),
                                  *(const v8bf*)(Ah + aoff + 16));
            const v16bf al = cat8(*(const v8bf*)(Al + aoff),
                                  *(const v8bf*)(Al + aoff + 16));
#pragma unroll
            for (int t = 0; t < 4; ++t) {
                // B fragment: lane half h holds k = 8h+[0..7], 16+8h+[0..7], col = lane&15
                const int boff = (wn * 64 + t * 16 + laneM) * PAD_ROW + half * 8;
                const v16bf bh = cat8(*(const v8bf*)(Bh + boff),
                                      *(const v8bf*)(Bh + boff + 16));
                const v16bf bl = cat8(*(const v8bf*)(Bl + boff),
                                      *(const v8bf*)(Bl + boff + 16));
                acc[t] = __builtin_amdgcn_wmma_f32_16x16x32_bf16(
                             false, ah, false, bh, (short)0, acc[t], false, false);
                acc[t] = __builtin_amdgcn_wmma_f32_16x16x32_bf16(
                             false, al, false, bh, (short)0, acc[t], false, false);
                acc[t] = __builtin_amdgcn_wmma_f32_16x16x32_bf16(
                             false, ah, false, bl, (short)0, acc[t], false, false);
            }
        }
        __syncthreads();
    }

    // Epilogue: bias + exact GELU. C/D layout: VGPR r -> row r (lanes 0-15) /
    // row r+8 (lanes 16-31), col = lane&15.
#pragma unroll
    for (int t = 0; t < 4; ++t) {
        const int ncol = n0 + wn * 64 + t * 16 + laneM;
        const float bb = be[ncol];
#pragma unroll
        for (int r = 0; r < 8; ++r) {
            const int m = m0 + wm * 16 + r + half * 8;
            float v = acc[t][r] + bb;
            if (applyGelu)
                v = 0.5f * v * (1.0f + erff(v * 0.70710678118654752f));
            Ce[(size_t)m * N + ncol] = v;
        }
    }
}

// -------------------------------------------------------------- combine -----

__global__ void moe_combine_kernel(const float* __restrict__ y,
                                   const float* __restrict__ fb,
                                   const int* __restrict__ used,
                                   const int* __restrict__ token_slots,
                                   float* __restrict__ out) {
    int i = blockIdx.x * blockDim.x + threadIdx.x;
    int t = i >> 9;
    int d = i & 511;
    int u = used[t];
    float r;
    if (u > 0) {
        int s0 = token_slots[2 * t];
        float s = y[(size_t)s0 * D_MODEL + d];
        if (u > 1) s += y[(size_t)token_slots[2 * t + 1] * D_MODEL + d];
        r = s / (float)u;
    } else {
        r = fb[(size_t)t * D_MODEL + d];
    }
    out[i] = r;
}

// ---------------------------------------------------------------- launch ----

extern "C" void kernel_launch(void* const* d_in, const int* in_sizes, int n_in,
                              void* d_out, int out_size, void* d_ws, size_t ws_size,
                              hipStream_t stream) {
    (void)in_sizes; (void)n_in; (void)out_size; (void)ws_size;

    const float* x   = (const float*)d_in[0];
    const float* W1  = (const float*)d_in[1];
    const float* b1  = (const float*)d_in[2];
    const float* W2  = (const float*)d_in[3];
    const float* b2  = (const float*)d_in[4];
    const float* Wf1 = (const float*)d_in[5];
    const float* bf1 = (const float*)d_in[6];
    const float* Wf2 = (const float*)d_in[7];
    const float* bf2 = (const float*)d_in[8];
    const int* routes = (const int*)d_in[9];
    float* out = (float*)d_out;

    char* ws = (char*)d_ws;
    size_t off = 0;
    float* xg = (float*)(ws + off); off += (size_t)SLOTS * D_MODEL * 4;   // 10.5 MB
    float* H  = (float*)(ws + off); off += (size_t)SLOTS * D_FF * 4;      // 42 MB (reused as Hf)
    float* y  = (float*)(ws + off); off += (size_t)SLOTS * D_MODEL * 4;   // 10.5 MB
    float* fb = (float*)(ws + off); off += (size_t)BTOK * D_MODEL * 4;    // 4.2 MB
    int* tok_of_slot = (int*)(ws + off); off += (size_t)SLOTS * 4;
    int* used        = (int*)(ws + off); off += (size_t)BTOK * 4;
    int* token_slots = (int*)(ws + off); off += (size_t)2 * BTOK * 4;
    float* Hf = H;   // expert H is dead before fallback layer1 runs

    moe_init_kernel<<<(SLOTS + 255) / 256, 256, 0, stream>>>(tok_of_slot, used, token_slots);
    moe_route_kernel<<<1, 32, 0, stream>>>(routes, tok_of_slot, used, token_slots);
    moe_gather_kernel<<<(SLOTS * D_MODEL) / 256, 256, 0, stream>>>(x, tok_of_slot, xg);

    // Expert layer 1: [320x512] @ [512x2048] + b1, GELU   (per expert, grid.z)
    moe_wmma_gemm_kernel<<<dim3((CAP / BLK_M) * (D_FF / BLK_N), 1, NE), 256, 0, stream>>>(
        xg, (long long)CAP * D_MODEL, W1, (long long)D_MODEL * D_FF,
        b1, D_FF, H, (long long)CAP * D_FF,
        D_FF, D_MODEL, D_FF / BLK_N, 1);
    // Expert layer 2: [320x2048] @ [2048x512] + b2
    moe_wmma_gemm_kernel<<<dim3((CAP / BLK_M) * (D_MODEL / BLK_N), 1, NE), 256, 0, stream>>>(
        H, (long long)CAP * D_FF, W2, (long long)D_FF * D_MODEL,
        b2, D_MODEL, y, (long long)CAP * D_MODEL,
        D_MODEL, D_FF, D_MODEL / BLK_N, 0);
    // Fallback layer 1: [2048x512] @ [512x2048] + bf1, GELU
    moe_wmma_gemm_kernel<<<dim3((BTOK / BLK_M) * (D_FF / BLK_N), 1, 1), 256, 0, stream>>>(
        x, 0, Wf1, 0, bf1, 0, Hf, 0,
        D_FF, D_MODEL, D_FF / BLK_N, 1);
    // Fallback layer 2: [2048x2048] @ [2048x512] + bf2
    moe_wmma_gemm_kernel<<<dim3((BTOK / BLK_M) * (D_MODEL / BLK_N), 1, 1), 256, 0, stream>>>(
        Hf, 0, Wf2, 0, bf2, 0, fb, 0,
        D_MODEL, D_FF, D_MODEL / BLK_N, 0);

    moe_combine_kernel<<<(BTOK * D_MODEL) / 256, 256, 0, stream>>>(
        y, fb, used, token_slots, out);
}
